// RelationshipAttention_90074054131968
// MI455X (gfx1250) — matrix operand order repaired and checked by reading, not compiled, not run on hardware
//
#include <hip/hip_runtime.h>
#include <hip/hip_bf16.h>
#include <stdint.h>

typedef __attribute__((ext_vector_type(16))) __bf16 v16bf;
typedef __attribute__((ext_vector_type(8)))  __bf16 v8bf;
typedef __attribute__((ext_vector_type(4)))  __bf16 v4bf;
typedef __attribute__((ext_vector_type(8)))  float  v8f;

#define BATCH 8
#define NSEQ  2048
#define DDIM  768
#define KTOP  10
#define RTOP  5
#define ROWS_PB 32          // rows per block (2 waves x 16)
#define CHUNK  256          // K-dim staging chunk (bf16 elems)

// ---------------------------------------------------------------------------
// Kernel 1: raw diagonal dot products  rawdiag[b*N+i] = dot(q_i, k_i)  (f32)
// ---------------------------------------------------------------------------
__global__ __launch_bounds__(256) void diag_kernel(const float* __restrict__ q,
                                                   const float* __restrict__ k,
                                                   float* __restrict__ rawdiag) {
    int row  = blockIdx.x * 8 + (threadIdx.x >> 5);   // 8 waves per block
    int lane = threadIdx.x & 31;
    const float* qp = q + (size_t)row * DDIM;
    const float* kp = k + (size_t)row * DDIM;
    float s = 0.f;
    #pragma unroll 4
    for (int t = lane; t < DDIM; t += 32) s += qp[t] * kp[t];
    #pragma unroll
    for (int off = 16; off; off >>= 1) s += __shfl_xor(s, off, 32);
    if (lane == 0) rawdiag[row] = s;
}

// ---------------------------------------------------------------------------
// Kernel 1b: one-shot f32 -> bf16 conversion (amortizes conversion, halves
// the bytes the WMMA kernel streams from L2).
// ---------------------------------------------------------------------------
__global__ __launch_bounds__(256) void convert_kernel(const float* __restrict__ src,
                                                      __bf16* __restrict__ dst) {
    size_t i = ((size_t)blockIdx.x * 256 + threadIdx.x) * 8;
    const float4* s4 = reinterpret_cast<const float4*>(src + i);
    float4 a = s4[0], b = s4[1];
    v8bf o;
    o[0] = (__bf16)a.x; o[1] = (__bf16)a.y; o[2] = (__bf16)a.z; o[3] = (__bf16)a.w;
    o[4] = (__bf16)b.x; o[5] = (__bf16)b.y; o[6] = (__bf16)b.z; o[7] = (__bf16)b.w;
    *reinterpret_cast<v8bf*>(dst + i) = o;
}

// ---------------------------------------------------------------------------
// Shared WMMA / online-softmax helpers (used by both stats variants)
// ---------------------------------------------------------------------------
__device__ __forceinline__ v8f wmma_chunk(const __bf16* Qs, const __bf16* Ks,
                                          int wv, int lane, int ch, v8f acc) {
    const int half = lane >> 4;
    const int aRow = lane & 15;
    #pragma unroll
    for (int kk = 0; kk < CHUNK; kk += 32) {
        // A fragment: 16x32 bf16. lane<16: K=kk+0..7,kk+16..23;
        // lane>=16: K=kk+8..15,kk+24..31. Row = wv*16 + (lane&15).
        const __bf16* ap = &Qs[(wv * 16 + aRow) * DDIM + ch * CHUNK + kk + half * 8];
        v8bf alo = *reinterpret_cast<const v8bf*>(ap);
        v8bf ahi = *reinterpret_cast<const v8bf*>(ap + 16);
        v16bf a = __builtin_shufflevector(alo, ahi,
            0,1,2,3,4,5,6,7,8,9,10,11,12,13,14,15);
        // B fragment: 32x16 bf16. col n = lane&15; K = half*8 + 0..7, 16 + half*8 + 0..7.
        const __bf16* bp = &Ks[(lane & 15) * CHUNK + kk + half * 8];
        v8bf blo = *reinterpret_cast<const v8bf*>(bp);
        v8bf bhi = *reinterpret_cast<const v8bf*>(bp + 16);
        v16bf b = __builtin_shufflevector(blo, bhi,
            0,1,2,3,4,5,6,7,8,9,10,11,12,13,14,15);
        acc = __builtin_amdgcn_wmma_f32_16x16x32_bf16(
            false, a, false, b, (short)0, acc, false, false);
    }
    return acc;
}

__device__ __forceinline__ void online_merge(const v8f& acc, float rm[8], float rl[8]) {
    #pragma unroll
    for (int r = 0; r < 8; ++r) {
        float v  = acc[r];
        float nm = fmaxf(rm[r], v);
        rl[r] = rl[r] * __expf(rm[r] - nm) + __expf(v - nm);
        rm[r] = nm;
    }
}

__device__ __forceinline__ void stats_epilogue(float rm[8], float rl[8],
                                               int lane, int wv, int half,
                                               int batch, int rowBase,
                                               const float* __restrict__ rawdiag,
                                               float* __restrict__ mw,
                                               float* __restrict__ lw,
                                               float* __restrict__ conf) {
    // cross-lane (16-lane) merge of (m,l) pairs
    #pragma unroll
    for (int off = 1; off < 16; off <<= 1) {
        #pragma unroll
        for (int r = 0; r < 8; ++r) {
            float om = __shfl_xor(rm[r], off, 32);
            float ol = __shfl_xor(rl[r], off, 32);
            float nm = fmaxf(rm[r], om);
            rl[r] = rl[r] * __expf(rm[r] - nm) + ol * __expf(om - nm);
            rm[r] = nm;
        }
    }
    if ((lane & 15) == 0) {
        #pragma unroll
        for (int r = 0; r < 8; ++r) {
            int row = batch * NSEQ + rowBase + wv * 16 + r + 8 * half;
            mw[row] = rm[r];
            lw[row] = rl[r];
            conf[row] = __expf(rawdiag[row] - rm[r]) / rl[r];
        }
    }
}

// ---------------------------------------------------------------------------
// Kernel 2a (fast path): streaming row softmax stats via WMMA; bf16 inputs
// staged straight into LDS with CDNA5 async-to-LDS DMA (ASYNCcnt tracked).
// ---------------------------------------------------------------------------
__global__ __launch_bounds__(64) void stats_kernel_bf16(const __bf16* __restrict__ qbf,
                                                        const __bf16* __restrict__ kbf,
                                                        const float* __restrict__ rawdiag,
                                                        float* __restrict__ mw,
                                                        float* __restrict__ lw,
                                                        float* __restrict__ conf) {
    __shared__ __bf16 Qs[ROWS_PB * DDIM];   // 48 KB
    __shared__ __bf16 Ks[16 * CHUNK];       // 8 KB

    const int tid   = threadIdx.x;
    const int lane  = tid & 31;
    const int wv    = tid >> 5;
    const int half  = lane >> 4;
    const int bid   = blockIdx.x;
    const int batch = bid >> 6;             // / (N/ROWS_PB = 64)
    const int rowBase = (bid & 63) * ROWS_PB;

    const uint32_t qsBase = (uint32_t)(uintptr_t)(&Qs[0]);
    const uint32_t ksBase = (uint32_t)(uintptr_t)(&Ks[0]);

    // --- async-stage Q rows: contiguous 32x768 bf16 = 3072 x 16B units ---
    {
        const __bf16* qg = qbf + (size_t)(batch * NSEQ + rowBase) * DDIM;
        #pragma unroll 8
        for (int i = tid; i < ROWS_PB * DDIM / 8; i += 64) {
            uint64_t ga = (uint64_t)(uintptr_t)(qg + i * 8);
            uint32_t la = qsBase + (uint32_t)i * 16u;
            asm volatile("global_load_async_to_lds_b128 %0, %1, off"
                         :: "v"(la), "v"(ga) : "memory");
        }
        asm volatile("s_wait_asynccnt 0" ::: "memory");
    }
    __syncthreads();

    float rm[8], rl[8];
    #pragma unroll
    for (int r = 0; r < 8; ++r) { rm[r] = -__builtin_inff(); rl[r] = 0.f; }

    for (int ct = 0; ct < NSEQ / 16; ++ct) {
        v8f acc = {0.f, 0.f, 0.f, 0.f, 0.f, 0.f, 0.f, 0.f};
        const int colBase = ct * 16;
        for (int ch = 0; ch < DDIM / CHUNK; ++ch) {
            __syncthreads();   // previous chunk fully consumed
            // --- async-stage K tile chunk: 16 cols x 256 bf16 = 512 x 16B ---
            #pragma unroll
            for (int i = tid; i < 16 * CHUNK / 8; i += 64) {
                int r = i >> 5;          // 32 x 16B units per row
                int c = i & 31;
                const __bf16* src = kbf + (size_t)(batch * NSEQ + colBase + r) * DDIM
                                        + ch * CHUNK + c * 8;
                uint32_t la = ksBase + (uint32_t)(r * CHUNK + c * 8) * 2u;
                asm volatile("global_load_async_to_lds_b128 %0, %1, off"
                             :: "v"(la), "v"((uint64_t)(uintptr_t)src) : "memory");
            }
            asm volatile("s_wait_asynccnt 0" ::: "memory");
            __syncthreads();
            acc = wmma_chunk(Qs, Ks, wv, lane, ch, acc);
        }
        online_merge(acc, rm, rl);
    }
    stats_epilogue(rm, rl, lane, wv, half, batch, rowBase, rawdiag, mw, lw, conf);
}

// ---------------------------------------------------------------------------
// Kernel 2b (fallback if workspace too small for bf16 mirrors): stage f32
// from global, convert to bf16 on the fly.
// ---------------------------------------------------------------------------
__global__ __launch_bounds__(64) void stats_kernel_f32(const float* __restrict__ q,
                                                       const float* __restrict__ k,
                                                       const float* __restrict__ rawdiag,
                                                       float* __restrict__ mw,
                                                       float* __restrict__ lw,
                                                       float* __restrict__ conf) {
    __shared__ __bf16 Qs[ROWS_PB * DDIM];   // 48 KB
    __shared__ __bf16 Ks[16 * CHUNK];       // 8 KB

    const int tid   = threadIdx.x;
    const int lane  = tid & 31;
    const int wv    = tid >> 5;
    const int half  = lane >> 4;
    const int bid   = blockIdx.x;
    const int batch = bid >> 6;
    const int rowBase = (bid & 63) * ROWS_PB;

    const float4* q4 = reinterpret_cast<const float4*>(
        q + (size_t)(batch * NSEQ + rowBase) * DDIM);
    v4bf* Qs4 = reinterpret_cast<v4bf*>(Qs);
    #pragma unroll 4
    for (int i = tid; i < ROWS_PB * DDIM / 4; i += 64) {
        float4 f = q4[i];
        v4bf o;
        o[0] = (__bf16)f.x; o[1] = (__bf16)f.y;
        o[2] = (__bf16)f.z; o[3] = (__bf16)f.w;
        Qs4[i] = o;
    }
    __syncthreads();

    float rm[8], rl[8];
    #pragma unroll
    for (int r = 0; r < 8; ++r) { rm[r] = -__builtin_inff(); rl[r] = 0.f; }

    const float4* k4 = reinterpret_cast<const float4*>(k);
    for (int ct = 0; ct < NSEQ / 16; ++ct) {
        v8f acc = {0.f, 0.f, 0.f, 0.f, 0.f, 0.f, 0.f, 0.f};
        const int colBase = ct * 16;
        for (int ch = 0; ch < DDIM / CHUNK; ++ch) {
            __syncthreads();
            v4bf* Ks4 = reinterpret_cast<v4bf*>(Ks);
            #pragma unroll 4
            for (int i = tid; i < 16 * CHUNK / 4; i += 64) {
                int r  = i >> 6;
                int c4 = i & 63;
                float4 f = k4[(size_t)(batch * NSEQ + colBase + r) * (DDIM / 4)
                              + ch * (CHUNK / 4) + c4];
                v4bf o;
                o[0] = (__bf16)f.x; o[1] = (__bf16)f.y;
                o[2] = (__bf16)f.z; o[3] = (__bf16)f.w;
                Ks4[i] = o;
            }
            __syncthreads();
            acc = wmma_chunk(Qs, Ks, wv, lane, ch, acc);
        }
        online_merge(acc, rm, rl);
    }
    stats_epilogue(rm, rl, lane, wv, half, batch, rowBase, rawdiag, mw, lw, conf);
}

// ---------------------------------------------------------------------------
// Kernel 3: per-batch top-K instances, top-R relationships, gather outputs.
// ---------------------------------------------------------------------------
__global__ __launch_bounds__(256) void finalize_kernel(const float* __restrict__ q,
                                                       const float* __restrict__ k,
                                                       const float* __restrict__ mw,
                                                       const float* __restrict__ lw,
                                                       const float* __restrict__ conf,
                                                       float* __restrict__ out) {
    __shared__ int   chosen[KTOP];
    __shared__ float redV[256];
    __shared__ int   redI[256];
    __shared__ float rel[KTOP * KTOP];
    __shared__ int   relIdx[KTOP][RTOP];

    const int b = blockIdx.x;
    const int tid = threadIdx.x;
    const float* cb = conf + (size_t)b * NSEQ;

    // ---- top-10 of conf via repeated parallel argmax (stable tie-break) ----
    for (int kq = 0; kq < KTOP; ++kq) {
        float bv = -__builtin_inff(); int bi = NSEQ;
        for (int j = tid; j < NSEQ; j += 256) {
            bool skip = false;
            for (int t = 0; t < kq; ++t) if (chosen[t] == j) skip = true;
            if (skip) continue;
            float v = cb[j];
            if (v > bv || (v == bv && j < bi)) { bv = v; bi = j; }
        }
        redV[tid] = bv; redI[tid] = bi;
        __syncthreads();
        for (int s = 128; s > 0; s >>= 1) {
            if (tid < s) {
                float ov = redV[tid + s]; int oi = redI[tid + s];
                if (ov > redV[tid] || (ov == redV[tid] && oi < redI[tid])) {
                    redV[tid] = ov; redI[tid] = oi;
                }
            }
            __syncthreads();
        }
        if (tid == 0) chosen[kq] = redI[0];
        __syncthreads();
    }
    if (tid == 0) {   // ascending sort
        for (int a = 1; a < KTOP; ++a) {
            int v = chosen[a]; int t = a - 1;
            while (t >= 0 && chosen[t] > v) { chosen[t + 1] = chosen[t]; --t; }
            chosen[t + 1] = v;
        }
    }
    __syncthreads();

    // ---- rel_scores[i][j] = exp(q_I . k_J - m_I) / l_I  (f32 recompute) ----
    const int wv = tid >> 5, lane = tid & 31;
    for (int p = wv; p < KTOP * KTOP; p += 8) {
        int I = chosen[p / KTOP], J = chosen[p % KTOP];
        const float* qp = q + (size_t)(b * NSEQ + I) * DDIM;
        const float* kp = k + (size_t)(b * NSEQ + J) * DDIM;
        float s = 0.f;
        for (int t = lane; t < DDIM; t += 32) s += qp[t] * kp[t];
        #pragma unroll
        for (int off = 16; off; off >>= 1) s += __shfl_xor(s, off, 32);
        if (lane == 0) {
            int g = b * NSEQ + I;
            rel[p] = __expf(s - mw[g]) / lw[g];
        }
    }
    __syncthreads();

    // ---- top-5 per row, ascending local column indices ----
    if (tid < KTOP) {
        int i = tid;
        int pk[RTOP];
        for (int r = 0; r < RTOP; ++r) {
            float bv = -__builtin_inff(); int bi = KTOP;
            for (int j = 0; j < KTOP; ++j) {
                bool used = false;
                for (int t = 0; t < r; ++t) if (pk[t] == j) used = true;
                if (used) continue;
                float v = rel[i * KTOP + j];
                if (v > bv || (v == bv && j < bi)) { bv = v; bi = j; }
            }
            pk[r] = bi;
        }
        for (int a = 1; a < RTOP; ++a) {
            int v = pk[a]; int t = a - 1;
            while (t >= 0 && pk[t] > v) { pk[t + 1] = pk[t]; --t; }
            pk[t + 1] = v;
        }
        for (int r = 0; r < RTOP; ++r) relIdx[i][r] = pk[r];
    }
    __syncthreads();

    // ---- gather subject/object/relationship embeddings ----
    const int PAIRS = KTOP * RTOP;                // 50
    for (int idx = tid; idx < PAIRS * DDIM; idx += 256) {
        int p = idx / DDIM, d = idx - p * DDIM;
        int i = p / RTOP, rr = p - i * RTOP;
        int subj = chosen[i];
        int obj  = chosen[relIdx[i][rr]];
        float sv = q[(size_t)(b * NSEQ + subj) * DDIM + d];
        float ov = q[(size_t)(b * NSEQ + obj)  * DDIM + d];
        size_t o = (size_t)(b * PAIRS + p) * DDIM + d;
        out[o]                                    = sv;        // subject
        out[(size_t)BATCH * PAIRS * DDIM + o]     = ov;        // object
        out[(size_t)2 * BATCH * PAIRS * DDIM + o] = sv + ov;   // relationship
    }
}

// ---------------------------------------------------------------------------
extern "C" void kernel_launch(void* const* d_in, const int* in_sizes, int n_in,
                              void* d_out, int out_size, void* d_ws, size_t ws_size,
                              hipStream_t stream) {
    const float* q = (const float*)d_in[0];
    const float* k = (const float*)d_in[1];
    // d_in[2]=top_k_instances(10), d_in[3]=top_k_relationships(5): compile-time here.

    const size_t BND      = (size_t)BATCH * NSEQ * DDIM;      // 12.58M elems
    const size_t statsFl  = (size_t)4 * BATCH * NSEQ;         // 4 x B*N f32
    const size_t needFast = 2 * BND * sizeof(__bf16) + statsFl * sizeof(float);

    if (ws_size >= needFast) {
        // Fast path: bf16 mirrors + async-to-LDS staging.
        __bf16* qbf = (__bf16*)d_ws;
        __bf16* kbf = qbf + BND;
        float*  st  = (float*)(kbf + BND);
        float* rawdiag = st;
        float* mw   = st + BATCH * NSEQ;
        float* lw   = st + 2 * BATCH * NSEQ;
        float* conf = st + 3 * BATCH * NSEQ;

        convert_kernel<<<(int)(BND / 8 / 256), 256, 0, stream>>>(q, qbf);
        convert_kernel<<<(int)(BND / 8 / 256), 256, 0, stream>>>(k, kbf);
        diag_kernel<<<BATCH * NSEQ / 8, 256, 0, stream>>>(q, k, rawdiag);
        stats_kernel_bf16<<<BATCH * (NSEQ / ROWS_PB), 64, 0, stream>>>(
            qbf, kbf, rawdiag, mw, lw, conf);
        finalize_kernel<<<BATCH, 256, 0, stream>>>(q, k, mw, lw, conf, (float*)d_out);
    } else {
        // Fallback: convert-in-kernel staging, only 256 KB of workspace.
        float* ws      = (float*)d_ws;
        float* rawdiag = ws;
        float* mw      = ws + BATCH * NSEQ;
        float* lw      = ws + 2 * BATCH * NSEQ;
        float* conf    = ws + 3 * BATCH * NSEQ;

        diag_kernel<<<BATCH * NSEQ / 8, 256, 0, stream>>>(q, k, rawdiag);
        stats_kernel_f32<<<BATCH * (NSEQ / ROWS_PB), 64, 0, stream>>>(
            q, k, rawdiag, mw, lw, conf);
        finalize_kernel<<<BATCH, 256, 0, stream>>>(q, k, mw, lw, conf, (float*)d_out);
    }

    (void)in_sizes; (void)n_in; (void)out_size;
}
